// LSTMClassifier_47837345743227
// MI455X (gfx1250) — matrix operand order repaired and checked by reading, not compiled, not run on hardware
//
#include <hip/hip_runtime.h>

// ---------- CDNA5 WMMA types ----------
typedef __attribute__((ext_vector_type(16))) __bf16 v16bf;
typedef __attribute__((ext_vector_type(8)))  float  v8f;
typedef __attribute__((ext_vector_type(2)))  float  v2f;

union FragBF {
    v16bf v;
    unsigned short u[16];
    unsigned int   p[8];
    uint4          q[2];   // two 16B halves for vectorized LDS loads
};

static __device__ __forceinline__ unsigned short f2bf(float f) {
    unsigned u = __builtin_bit_cast(unsigned, f);
    u += 0x7FFFu + ((u >> 16) & 1u);          // round-to-nearest-even
    return (unsigned short)(u >> 16);
}
static __device__ __forceinline__ float bf2f(unsigned short h) {
    return __builtin_bit_cast(float, (unsigned)h << 16);
}
static __device__ __forceinline__ float sigmoidf(float x) {
    return 1.0f / (1.0f + __expf(-x));
}

// Problem constants
#define BATCH 64
#define TIN   768
#define DIN   128
#define EDIM  64
#define HDIM  128
#define TC    256
#define KK    384
#define NCOL  (BATCH*TC)   // 16384
#define ODIM  10
#define EPS_PROX 5.0f

// ---------------------------------------------------------------------------
// Kernel 1: conv weights fp32 [E][D][K] -> bf16 [E][384]
// ---------------------------------------------------------------------------
__global__ void pack_w_kernel(const float* __restrict__ w, unsigned short* __restrict__ wc, int n) {
    int i = blockIdx.x * 256 + threadIdx.x;
    if (i < n) wc[i] = f2bf(w[i]);
}

// ---------------------------------------------------------------------------
// Kernel 2: L2 normalize over D + im2col-pack to bf16 (stride==kernel)
// ---------------------------------------------------------------------------
__global__ void l2norm_pack_kernel(const float* __restrict__ x, unsigned short* __restrict__ Xc) {
    int row = blockIdx.x;                 // b*768 + tin
    int d   = threadIdx.x;                // 0..127
    float v = x[(size_t)row * DIN + d];
    __shared__ float red[DIN];
    red[d] = v * v;
    __syncthreads();
    for (int s = 64; s; s >>= 1) {
        if (d < s) red[d] += red[d + s];
        __syncthreads();
    }
    float nrm = fmaxf(sqrtf(red[0]), 1e-12f);
    float xn  = v / nrm;
    int b  = row / TIN;
    int ti = row - b * TIN;
    int t  = ti / 3;
    int kt = ti - 3 * t;
    Xc[((size_t)(b * TC + t)) * KK + d * 3 + kt] = f2bf(xn);
}

// ---------------------------------------------------------------------------
// Kernel 3: conv GEMM via bf16 WMMA: emb[E][NCOL] = Wc @ Xc^T + conv_b
// ---------------------------------------------------------------------------
__global__ void conv_gemm_kernel(const unsigned short* __restrict__ Wc,
                                 const unsigned short* __restrict__ Xc,
                                 const float* __restrict__ conv_b,
                                 float* __restrict__ emb) {
    int lane = threadIdx.x & 31;
    int wave = threadIdx.x >> 5;
    int tile = blockIdx.x * 8 + wave;     // 0..4095
    int tm = tile & 3;
    int tn = tile >> 2;
    int rowl = lane & 15;
    int hi   = lane >> 4;

    int arow = tm * 16 + rowl;
    int brow = tn * 16 + rowl;
    v8f acc = {};
    for (int k0 = 0; k0 < KK; k0 += 32) {
        FragBF a, bm;
#pragma unroll
        for (int v = 0; v < 8; v++) {
            int base = (v < 4) ? (hi * 8 + 2 * v) : (16 + hi * 8 + 2 * (v - 4));
            a.p[v]  = *(const unsigned*)(Wc + (size_t)arow * KK + k0 + base);
            bm.p[v] = *(const unsigned*)(Xc + (size_t)brow * KK + k0 + base);
        }
        acc = __builtin_amdgcn_wmma_f32_16x16x32_bf16(false, a.v, false, bm.v,
                                                      (short)0, acc, false, false);
    }
#pragma unroll
    for (int r = 0; r < 8; r++) {
        int e = tm * 16 + r + 8 * hi;
        int n = tn * 16 + rowl;
        emb[(size_t)e * NCOL + n] = acc[r] + conv_b[e];
    }
}

// ---------------------------------------------------------------------------
// Kernel 4: BN batch stats per channel -> scale/shift
// ---------------------------------------------------------------------------
__global__ void bn_stats_kernel(const float* __restrict__ emb,
                                const float* __restrict__ gamma,
                                const float* __restrict__ beta,
                                float* __restrict__ stats) {
    int e = blockIdx.x, tid = threadIdx.x;
    float s1 = 0.f, s2 = 0.f;
    for (int n = tid; n < NCOL; n += 256) {
        float v = emb[(size_t)e * NCOL + n];
        s1 += v; s2 += v * v;
    }
    __shared__ float r1[256], r2[256];
    r1[tid] = s1; r2[tid] = s2;
    __syncthreads();
    for (int s = 128; s; s >>= 1) {
        if (tid < s) { r1[tid] += r1[tid + s]; r2[tid] += r2[tid + s]; }
        __syncthreads();
    }
    if (tid == 0) {
        float mean = r1[0] * (1.0f / NCOL);
        float var  = r2[0] * (1.0f / NCOL) - mean * mean;
        float sc   = gamma[e] * rsqrtf(var + 1e-5f);
        stats[2 * e]     = sc;
        stats[2 * e + 1] = beta[e] - mean * sc;
    }
}

// ---------------------------------------------------------------------------
// Kernel 5: BN + ReLU -> bf16 act in [t][b][e] layout
// ---------------------------------------------------------------------------
__global__ void bn_apply_kernel(const float* __restrict__ emb,
                                const float* __restrict__ stats,
                                unsigned short* __restrict__ act) {
    int idx = blockIdx.x * 256 + threadIdx.x;
    int e = idx >> 14;
    int n = idx & (NCOL - 1);
    float v = emb[idx];
    v = fmaxf(v * stats[2 * e] + stats[2 * e + 1], 0.f);
    int b = n >> 8;
    int t = n & 255;
    act[(((size_t)t * BATCH + b) * EDIM) + e] = f2bf(v);
}

// ---------------------------------------------------------------------------
// Kernel 6: persistent ProxLSTM. One WG (256 thr / 8 waves) per batch element.
//  - A = I + eps*G*G^T : bf16 WMMA (16x16x32), b128 LDS fragment loads
//  - blocked Cholesky NB=16: wave-synchronous diag factor, per-row L21
//    triangular solves, SYRK trailing update via fp32 WMMA (16x16x4, K=16)
//  - blocked triangular solves: wave0 shfl diag solve + 16-wide matvec sweeps
// ---------------------------------------------------------------------------
#define A_PITCH 132
#define G_PITCH 80
#define LSTM_SMEM_BYTES (HDIM*A_PITCH*4 + HDIM*G_PITCH*2 + 512*4 + EDIM*4 + 5*HDIM*4)

__global__ void prox_lstm_kernel(const unsigned short* __restrict__ act,
                                 const float* __restrict__ w_ih,
                                 const float* __restrict__ w_hh,
                                 const float* __restrict__ b_ih,
                                 const float* __restrict__ b_hh,
                                 const float* __restrict__ lin_w,
                                 const float* __restrict__ lin_b,
                                 float* __restrict__ out) {
    extern __shared__ char smem[];
    float* As = (float*)smem;                               // [128][132] fp32
    unsigned short* Gs = (unsigned short*)(As + HDIM * A_PITCH); // [128][80] bf16 (16B-aligned rows)
    float* z  = (float*)(Gs + HDIM * G_PITCH);              // [512]
    float* xs = z + 512;                                    // [64]
    float* hv = xs + EDIM;                                  // [128]
    float* cv = hv + HDIM;                                  // [128]
    float* sv = cv + HDIM;                                  // [128]
    float* ov = sv + HDIM;                                  // [128]
    float* yv = ov + HDIM;                                  // [128]

    const int tid  = threadIdx.x;
    const int b    = blockIdx.x;
    const int lane = tid & 31;
    const int wave = tid >> 5;
    const int rowl = lane & 15;
    const int hi   = lane >> 4;

    for (int i = tid; i < HDIM; i += 256) { hv[i] = 0.f; cv[i] = 0.f; }
    __syncthreads();

    for (int t = 0; t < TC; t++) {
        // ---- x_t into LDS ----
        if (tid < EDIM) xs[tid] = bf2f(act[(((size_t)t * BATCH + b) * EDIM) + tid]);
        __builtin_prefetch(w_hh + ((size_t)(tid * 2) << 7), 0, 0);
        __syncthreads();

        // ---- z = W_ih x + b_ih + W_hh h + b_hh (2 rows / thread) ----
#pragma unroll
        for (int r = 0; r < 2; r++) {
            int j = tid * 2 + r;
            float accz = b_ih[j] + b_hh[j];
            const float* wi = w_ih + (size_t)j * EDIM;
#pragma unroll 4
            for (int k = 0; k < EDIM; k++) accz = fmaf(wi[k], xs[k], accz);
            const float* wh = w_hh + (size_t)j * HDIM;
#pragma unroll 4
            for (int k = 0; k < HDIM; k++) accz = fmaf(wh[k], hv[k], accz);
            z[j] = accz;
        }
        __syncthreads();

        // ---- gates, pre-prox state s, Jacobian coefficients ----
        if (tid < HDIM) {
            float ig = sigmoidf(z[tid]);
            float fg = sigmoidf(z[tid + 128]);
            float gg = tanhf(z[tid + 256]);
            float og = sigmoidf(z[tid + 384]);
            float cold = cv[tid];
            sv[tid] = fg * cold + ig * gg;
            ov[tid] = og;
            z[tid]       = ig * (1.f - ig) * gg;    // ci
            z[tid + 128] = fg * (1.f - fg) * cold;  // cf
            z[tid + 256] = ig * (1.f - gg * gg);    // cg
        }
        __syncthreads();

        // ---- G[h][e] = ci*Wi + cf*Wf + cg*Wg  (bf16 in LDS) ----
        for (int idx = tid; idx < HDIM * EDIM; idx += 256) {
            int hr = idx >> 6, e = idx & 63;
            float g = z[hr]       * w_ih[(size_t)hr * EDIM + e]
                    + z[hr + 128] * w_ih[(size_t)(hr + 128) * EDIM + e]
                    + z[hr + 256] * w_ih[(size_t)(hr + 256) * EDIM + e];
            Gs[hr * G_PITCH + e] = f2bf(g);
        }
        __syncthreads();

        // ---- A = I + eps * G G^T via v_wmma_f32_16x16x32_bf16 ----
        {
            int cn = wave * 16;            // wave owns one 16-wide column strip
            for (int tm = 0; tm < 8; tm++) {
                int rm = tm * 16;
                int arow = rm + rowl;
                int brow = cn + rowl;
                v8f acc = {};
#pragma unroll
                for (int k0 = 0; k0 < EDIM; k0 += 32) {
                    FragBF a, bm;
                    // A frag: two contiguous 16B runs per lane (K = hi*8..+7, 16+hi*8..+7)
                    a.q[0] = *(const uint4*)(Gs + arow * G_PITCH + k0 + hi * 8);
                    a.q[1] = *(const uint4*)(Gs + arow * G_PITCH + k0 + 16 + hi * 8);
                    // B frag: two contiguous 16B runs per lane (K = hi*8..+7, 16+hi*8..+7)
                    bm.q[0] = *(const uint4*)(Gs + brow * G_PITCH + k0 + hi * 8);
                    bm.q[1] = *(const uint4*)(Gs + brow * G_PITCH + k0 + 16 + hi * 8);
                    acc = __builtin_amdgcn_wmma_f32_16x16x32_bf16(false, a.v, false, bm.v,
                                                                  (short)0, acc, false, false);
                }
#pragma unroll
                for (int r = 0; r < 8; r++) {
                    int m = rm + r + 8 * hi;
                    int n = cn + rowl;
                    As[m * A_PITCH + n] = EPS_PROX * acc[r] + ((m == n) ? 1.f : 0.f);
                }
            }
        }
        __syncthreads();

        // ================= blocked Cholesky, NB = 16 =================
        for (int p = 0; p < 8; p++) {
            int kb = p * 16;
            // --- 1) 16x16 diagonal block factor: wave 0, wave-synchronous ---
            if (wave == 0) {
                for (int k2 = 0; k2 < 16; k2++) {
                    int k = kb + k2;
                    if (lane == 0) As[k * A_PITCH + k] = sqrtf(As[k * A_PITCH + k]);
                    float dk = As[k * A_PITCH + k];
                    if (lane > k2 && lane < 16) As[(kb + lane) * A_PITCH + k] /= dk;
                    for (int e = lane; e < 256; e += 32) {
                        int i2 = e >> 4, j2 = e & 15;
                        if (j2 > k2 && j2 <= i2)
                            As[(kb + i2) * A_PITCH + kb + j2] -=
                                As[(kb + i2) * A_PITCH + k] * As[(kb + j2) * A_PITCH + k];
                    }
                }
            }
            __syncthreads();
            // --- 2) L21: independent per-row triangular solves (rows kb+16..127) ---
            for (int i = kb + 16 + tid; i < HDIM; i += 256) {
                float a[16];
#pragma unroll
                for (int j2 = 0; j2 < 16; j2++) a[j2] = As[i * A_PITCH + kb + j2];
#pragma unroll
                for (int k2 = 0; k2 < 16; k2++) {
                    float l = a[k2] / As[(kb + k2) * A_PITCH + kb + k2];
                    a[k2] = l;
#pragma unroll
                    for (int j2 = k2 + 1; j2 < 16; j2++)
                        a[j2] -= As[(kb + j2) * A_PITCH + kb + k2] * l;
                }
#pragma unroll
                for (int j2 = 0; j2 < 16; j2++) As[i * A_PITCH + kb + j2] = a[j2];
            }
            __syncthreads();
            // --- 3) SYRK trailing update via fp32 WMMA (K=16 as 4 x 16x16x4) ---
            int nb = 7 - p;                       // trailing block-rows
            int nblk = nb * (nb + 1) / 2;
            for (int blk = wave; blk < nblk; blk += 8) {
                int bi = 0;
                while ((bi + 1) * (bi + 2) / 2 <= blk) bi++;
                int bj = blk - bi * (bi + 1) / 2;
                int i0 = kb + 16 + bi * 16;
                int j0 = kb + 16 + bj * 16;
                v8f acc = {};
#pragma unroll
                for (int k4 = 0; k4 < 16; k4 += 4) {
                    // f32 A-frag (16x4): lane m=lane&15 holds K = k4 + 2*hi + {0,1}
                    v2f af = *(const v2f*)&As[(i0 + rowl) * A_PITCH + kb + k4 + 2 * hi];
                    v2f bf = *(const v2f*)&As[(j0 + rowl) * A_PITCH + kb + k4 + 2 * hi];
                    acc = __builtin_amdgcn_wmma_f32_16x16x4_f32(false, af, false, bf,
                                                                (short)0, acc, false, false);
                }
#pragma unroll
                for (int r = 0; r < 8; r++) {
                    int m = i0 + r + 8 * hi;
                    int n = j0 + rowl;
                    As[m * A_PITCH + n] -= acc[r];
                }
            }
            __syncthreads();
        }

        // ================= forward solve L y = s (blocked) =================
        for (int p = 0; p < 8; p++) {
            int kb = p * 16;
            if (wave == 0) {
                float acc = sv[kb + rowl];
                for (int k2 = 0; k2 < 16; k2++) {
                    float yk = __shfl(acc, k2, 32) / As[(kb + k2) * A_PITCH + kb + k2];
                    if (rowl == k2 && lane < 16) acc = yk;
                    if (rowl > k2) acc -= As[(kb + rowl) * A_PITCH + kb + k2] * yk;
                }
                if (lane < 16) yv[kb + rowl] = acc;
            }
            __syncthreads();
            for (int i = kb + 16 + tid; i < HDIM; i += 256) {
                float s = 0.f;
#pragma unroll
                for (int k2 = 0; k2 < 16; k2++) s += As[i * A_PITCH + kb + k2] * yv[kb + k2];
                sv[i] -= s;
            }
            __syncthreads();
        }
        // ================= backward solve L^T c = y (blocked) =================
        for (int p = 7; p >= 0; p--) {
            int kb = p * 16;
            if (wave == 0) {
                float acc = yv[kb + rowl];
                for (int k2 = 15; k2 >= 0; k2--) {
                    float ck = __shfl(acc, k2, 32) / As[(kb + k2) * A_PITCH + kb + k2];
                    if (rowl == k2 && lane < 16) acc = ck;
                    if (rowl < k2) acc -= As[(kb + k2) * A_PITCH + kb + rowl] * ck;
                }
                if (lane < 16) cv[kb + rowl] = acc;
            }
            __syncthreads();
            for (int i = tid; i < kb; i += 256) {
                float s = 0.f;
#pragma unroll
                for (int k2 = 0; k2 < 16; k2++) s += As[(kb + k2) * A_PITCH + i] * cv[kb + k2];
                yv[i] -= s;
            }
            __syncthreads();
        }

        // ---- h = o * tanh(c) ----
        if (tid < HDIM) hv[tid] = ov[tid] * tanhf(cv[tid]);
        __syncthreads();
    }

    // ---- final linear 128 -> 10 ----
    if (tid < ODIM) {
        float accl = lin_b[tid];
        for (int k = 0; k < HDIM; k++) accl = fmaf(lin_w[(size_t)tid * HDIM + k], hv[k], accl);
        out[b * ODIM + tid] = accl;
    }
}

// ---------------------------------------------------------------------------
extern "C" void kernel_launch(void* const* d_in, const int* in_sizes, int n_in,
                              void* d_out, int out_size, void* d_ws, size_t ws_size,
                              hipStream_t stream) {
    const float* inputs  = (const float*)d_in[0];   // [64,768,128]
    const float* conv_w  = (const float*)d_in[2];   // [64,128,3]
    const float* conv_b  = (const float*)d_in[3];   // [64]
    const float* bn_g    = (const float*)d_in[4];   // [64]
    const float* bn_b    = (const float*)d_in[5];   // [64]
    const float* w_ih    = (const float*)d_in[6];   // [512,64]
    const float* w_hh    = (const float*)d_in[7];   // [512,128]
    const float* b_ih    = (const float*)d_in[8];   // [512]
    const float* b_hh    = (const float*)d_in[9];   // [512]
    const float* lin_w   = (const float*)d_in[10];  // [10,128]
    const float* lin_b   = (const float*)d_in[11];  // [10]
    float* out = (float*)d_out;                     // [64,10]

    char* ws = (char*)d_ws;
    size_t off = 0;
    unsigned short* Xc = (unsigned short*)(ws + off); off += (size_t)NCOL * KK * 2;
    unsigned short* Wc = (unsigned short*)(ws + off); off += (size_t)EDIM * KK * 2;
    float* emb   = (float*)(ws + off);                off += (size_t)EDIM * NCOL * 4;
    float* stats = (float*)(ws + off);                off += 2 * EDIM * 4;
    off = (off + 255) & ~(size_t)255;
    unsigned short* act = (unsigned short*)(ws + off);

    pack_w_kernel<<<(EDIM * KK + 255) / 256, 256, 0, stream>>>(conv_w, Wc, EDIM * KK);
    l2norm_pack_kernel<<<BATCH * TIN, DIN, 0, stream>>>(inputs, Xc);
    conv_gemm_kernel<<<512, 256, 0, stream>>>(Wc, Xc, conv_b, emb);
    bn_stats_kernel<<<EDIM, 256, 0, stream>>>(emb, bn_g, bn_b, stats);
    bn_apply_kernel<<<(EDIM * NCOL) / 256, 256, 0, stream>>>(emb, stats, act);
    prox_lstm_kernel<<<BATCH, 256, LSTM_SMEM_BYTES, stream>>>(
        act, w_ih, w_hh, b_ih, b_hh, lin_w, lin_b, out);
}
